// VectorQuantizer_34359738640
// MI455X (gfx1250) — matrix operand order repaired and checked by reading, not compiled, not run on hardware
//
#include <hip/hip_runtime.h>

typedef __attribute__((ext_vector_type(16))) _Float16 v16h;
typedef __attribute__((ext_vector_type(8)))  _Float16 v8h;
typedef __attribute__((ext_vector_type(8)))  float    v8f;
typedef __attribute__((ext_vector_type(4)))  float    v4f;
typedef __attribute__((ext_vector_type(4)))  int      v4i;

#define BN_TOT 32768   // B*N
#define DDIM   256
#define KCODES 8192
#define NT     128     // codes per LDS tile (128*256*2B = 64KB), double buffered

// ---------------- workspace layout (bytes) ----------------
#define WS_ZN    0u                      // BN*D f16      = 16 MiB
#define WS_EN    16777216u               // K*D  f16      =  4 MiB
#define WS_IDX   20971520u               // BN   u32      = 128 KiB
#define WS_HIST  21102592u               // K    u32      =  32 KiB
#define WS_BSUM  21135360u               // BN   f32      = 128 KiB

// ---------------- output layout (floats) ----------------
#define OUT_ZQ    0u
#define OUT_IDX   8388608u
#define OUT_LOSS  8421376u
#define OUT_PERP  8421377u
#define OUT_USAGE 8421378u

#if defined(__HIP_DEVICE_COMPILE__) && __has_builtin(__builtin_amdgcn_global_load_async_to_lds_b128)
#define HAVE_ASYNC_LDS 1
#else
#define HAVE_ASYNC_LDS 0
#endif

#if HAVE_ASYNC_LDS
typedef __attribute__((address_space(1))) v4i glb_v4i;
typedef __attribute__((address_space(3))) v4i lds_v4i;
#endif

// Stage one NT x DDIM f16 tile (64KB) into LDS. 128 threads, 512B each.
__device__ __forceinline__ void tile_stage(const _Float16* __restrict__ g,
                                           _Float16* __restrict__ s, int tid) {
#if HAVE_ASYNC_LDS
    char* gp = const_cast<char*>((const char*)g);
    char* sp = (char*)s;
#pragma unroll
    for (int j = 0; j < 32; ++j) {
        int off = (tid + j * 128) * 16;
        __builtin_amdgcn_global_load_async_to_lds_b128(
            (glb_v4i*)(gp + off), (lds_v4i*)(sp + off), 0, 0);
    }
#else
    const v8h* gv = (const v8h*)g;
    v8h* sv = (v8h*)s;
#pragma unroll
    for (int j = 0; j < 32; ++j) {
        int c = tid + j * 128;
        sv[c] = gv[c];
    }
#endif
}

__device__ __forceinline__ void tile_wait() {
#if HAVE_ASYNC_LDS
#if __has_builtin(__builtin_amdgcn_s_wait_asynccnt)
    __builtin_amdgcn_s_wait_asynccnt(0);
#else
    asm volatile("s_wait_asynccnt 0" ::: "memory");
#endif
#endif
}

// =========================================================
// Zero the histogram
__global__ __launch_bounds__(256) void vq_init_kernel(unsigned* __restrict__ hist) {
    int i = blockIdx.x * 256 + threadIdx.x;
    if (i < KCODES) hist[i] = 0u;
}

// =========================================================
// L2-normalize rows (D=256) and convert to f16. One wave32 per row.
__global__ __launch_bounds__(256) void vq_norm_kernel(const float* __restrict__ src,
                                                      _Float16* __restrict__ dst,
                                                      int nrows) {
    const int wave = threadIdx.x >> 5;
    const int lane = threadIdx.x & 31;
    const int row  = blockIdx.x * 8 + wave;
    if (row >= nrows) return;

    const float* p = src + (size_t)row * DDIM + lane * 8;
    v4f a = *(const v4f*)p;
    v4f b = *(const v4f*)(p + 4);
    float vals[8];
#pragma unroll
    for (int i = 0; i < 4; ++i) { vals[i] = a[i]; vals[4 + i] = b[i]; }
    float ss = 0.f;
#pragma unroll
    for (int i = 0; i < 8; ++i) ss += vals[i] * vals[i];
#pragma unroll
    for (int m = 16; m >= 1; m >>= 1) ss += __shfl_xor(ss, m, 32);
    const float scale = 1.0f / fmaxf(sqrtf(ss), 1e-12f);

    v8h o;
#pragma unroll
    for (int i = 0; i < 8; ++i) o[i] = (_Float16)(vals[i] * scale);
    *(v8h*)(dst + (size_t)row * DDIM + lane * 8) = o;
}

// =========================================================
// Fused similarity GEMM (f16 WMMA, f32 acc) + running row-argmax.
// Block: 128 threads = 4 wave32; each wave owns 32 rows (two A fragment sets
// resident in VGPRs -> every B fragment read from LDS feeds TWO independent
// WMMA accumulator chains: halves LDS bandwidth per WMMA, doubles ILP).
// Codebook streamed through double-buffered 2x64KB LDS tiles (async-to-LDS
// when available, overlapped with compute).
__global__ __launch_bounds__(128) void vq_argmax_kernel(const _Float16* __restrict__ zn,
                                                        const _Float16* __restrict__ en,
                                                        unsigned* __restrict__ idx_out,
                                                        float* __restrict__ idxf_out) {
    __shared__ _Float16 sE[2][NT * DDIM];   // 128 KB

    const int tid  = threadIdx.x;
    const int wave = tid >> 5;
    const int lane = tid & 31;
    const int lo   = lane & 15;   // column / M-lane selector per ISA layouts
    const int hi   = lane >> 4;   // lane-half selector
    const int wave_row_base = blockIdx.x * 128 + wave * 32;

    // ---- Load A fragments once (16x32 A, contiguous k order, same as B):
    // lane element j: K = hi*16 + j
    v16h afragL[8], afragH[8];
#pragma unroll
    for (int kc = 0; kc < 8; ++kc) {
        const _Float16* pL = zn + (size_t)(wave_row_base + lo) * DDIM + kc * 32 + hi * 16;
        const _Float16* pH = pL + 16 * DDIM;
        v8h l0 = *(const v8h*)pL;
        v8h l1 = *(const v8h*)(pL + 8);
        v8h h0 = *(const v8h*)pH;
        v8h h1 = *(const v8h*)(pH + 8);
#pragma unroll
        for (int i = 0; i < 8; ++i) {
            afragL[kc][i] = l0[i]; afragL[kc][8 + i] = l1[i];
            afragH[kc][i] = h0[i]; afragH[kc][8 + i] = h1[i];
        }
    }

    float bestL[8], bestH[8];
    int   bidxL[8], bidxH[8];
#pragma unroll
    for (int i = 0; i < 8; ++i) {
        bestL[i] = -1.0e30f; bidxL[i] = 0;
        bestH[i] = -1.0e30f; bidxH[i] = 0;
    }

    // prologue: stage tile 0 into buffer 0
    tile_stage(en, &sE[0][0], tid);

    const int NTILES = KCODES / NT;
    for (int it = 0; it < NTILES; ++it) {
        tile_wait();          // my async loads into sE[it&1] complete
        __syncthreads();      // everyone's loads done; prev compute done with other buffer

        if (it + 1 < NTILES)  // overlap: stage next tile while computing this one
            tile_stage(en + (size_t)(it + 1) * NT * DDIM, &sE[(it + 1) & 1][0], tid);

        const _Float16* S = &sE[it & 1][0];
        for (int ns = 0; ns < 8; ++ns) {
            v8f accL = {}, accH = {};
            // B (32x16) layout: lane element j -> K = hi*16 + j, col N = lo.
            // B[k][n] = en[n][kbase + k] -> 16 contiguous halfs in LDS.
            const _Float16* bp = S + (size_t)(ns * 16 + lo) * DDIM + hi * 16;
#pragma unroll
            for (int kc = 0; kc < 8; ++kc) {
                v16h bfrag = *(const v16h*)(bp + kc * 32);
                accL = __builtin_amdgcn_wmma_f32_16x16x32_f16(
                    false, afragL[kc], false, bfrag, (short)0, accL, false, false);
                accH = __builtin_amdgcn_wmma_f32_16x16x32_f16(
                    false, afragH[kc], false, bfrag, (short)0, accH, false, false);
            }
            const int cand = it * NT + ns * 16 + lo;
#pragma unroll
            for (int i = 0; i < 8; ++i) {
                if (accL[i] > bestL[i]) { bestL[i] = accL[i]; bidxL[i] = cand; }
                if (accH[i] > bestH[i]) { bestH[i] = accH[i]; bidxH[i] = cand; }
            }
        }
    }

    // ---- cross-lane argmax over the 16-lane half (C layout: row = i + 8*hi)
#pragma unroll
    for (int m = 8; m >= 1; m >>= 1) {
#pragma unroll
        for (int i = 0; i < 8; ++i) {
            float pv = __shfl_xor(bestL[i], m, 32);
            int   pi = __shfl_xor(bidxL[i], m, 32);
            if (pv > bestL[i] || (pv == bestL[i] && pi < bidxL[i])) { bestL[i] = pv; bidxL[i] = pi; }
            pv = __shfl_xor(bestH[i], m, 32);
            pi = __shfl_xor(bidxH[i], m, 32);
            if (pv > bestH[i] || (pv == bestH[i] && pi < bidxH[i])) { bestH[i] = pv; bidxH[i] = pi; }
        }
    }
    if (lo == 0) {
#pragma unroll
        for (int i = 0; i < 8; ++i) {
            int rL = wave_row_base + hi * 8 + i;
            int rH = rL + 16;
            idx_out[rL]  = (unsigned)bidxL[i];
            idxf_out[rL] = (float)bidxL[i];
            idx_out[rH]  = (unsigned)bidxH[i];
            idxf_out[rH] = (float)bidxH[i];
        }
    }
}

// =========================================================
// Gather z_q, straight-through output, per-block SSE partial, histogram.
// One block (256 threads) per row.
__global__ __launch_bounds__(256) void vq_gather_kernel(const float* __restrict__ z_e,
                                                        const float* __restrict__ cb,
                                                        const unsigned* __restrict__ idx,
                                                        float* __restrict__ out_zq,
                                                        unsigned* __restrict__ hist,
                                                        float* __restrict__ bsum) {
    const int row = blockIdx.x;
    const int d   = threadIdx.x;
    const unsigned id = idx[row];

    const float ze = z_e[(size_t)row * DDIM + d];
    const float q  = cb[(size_t)id * DDIM + d];
    out_zq[(size_t)row * DDIM + d] = ze + (q - ze);   // straight-through

    float diff = q - ze;
    float sq = diff * diff;
#pragma unroll
    for (int m = 16; m >= 1; m >>= 1) sq += __shfl_xor(sq, m, 32);

    __shared__ float ssum[8];
    const int wave = threadIdx.x >> 5, lane = threadIdx.x & 31;
    if (lane == 0) ssum[wave] = sq;
    __syncthreads();
    if (threadIdx.x == 0) {
        float t = 0.f;
#pragma unroll
        for (int i = 0; i < 8; ++i) t += ssum[i];
        bsum[row] = t;
        atomicAdd(&hist[id], 1u);   // integer atomic: deterministic
    }
}

// =========================================================
// Entropy / perplexity / usage / loss. One block.
__global__ __launch_bounds__(256) void vq_finalize_kernel(const unsigned* __restrict__ hist,
                                                          const float* __restrict__ bsum,
                                                          float* __restrict__ out) {
    __shared__ float s1[256];
    __shared__ float s2[256];
    const int t = threadIdx.x;

    // total count + nonzero bins
    float total = 0.f, nz = 0.f;
    for (int j = t; j < KCODES; j += 256) {
        float h = (float)hist[j];
        total += h;
        nz += (h > 0.f) ? 1.f : 0.f;
    }
    s1[t] = total; s2[t] = nz;
    __syncthreads();
    for (int s = 128; s > 0; s >>= 1) { if (t < s) { s1[t] += s1[t + s]; s2[t] += s2[t + s]; } __syncthreads(); }
    const float tot = s1[0], nzc = s2[0];
    __syncthreads();

    // entropy
    float ent = 0.f;
    for (int j = t; j < KCODES; j += 256) {
        float p = (float)hist[j] / (tot + 1e-12f);
        ent -= p * logf(p + 1e-12f);
    }
    s1[t] = ent;
    // sse (deterministic tree over per-row partials)
    float sse = 0.f;
    for (int j = t; j < BN_TOT; j += 256) sse += bsum[j];
    s2[t] = sse;
    __syncthreads();
    for (int s = 128; s > 0; s >>= 1) { if (t < s) { s1[t] += s1[t + s]; s2[t] += s2[t + s]; } __syncthreads(); }

    if (t == 0) {
        out[OUT_LOSS]  = 1.25f * s2[0] / 8388608.0f;   // (1+beta)*MSE
        out[OUT_PERP]  = expf(s1[0]);
        out[OUT_USAGE] = nzc / (float)KCODES;
    }
}

// =========================================================
extern "C" void kernel_launch(void* const* d_in, const int* in_sizes, int n_in,
                              void* d_out, int out_size, void* d_ws, size_t ws_size,
                              hipStream_t stream) {
    const float* z_e = (const float*)d_in[0];   // [8,4096,256] f32
    const float* cb  = (const float*)d_in[1];   // [8192,256]   f32
    float* out = (float*)d_out;
    char*  ws  = (char*)d_ws;

    _Float16* zn   = (_Float16*)(ws + WS_ZN);
    _Float16* en   = (_Float16*)(ws + WS_EN);
    unsigned* idx  = (unsigned*)(ws + WS_IDX);
    unsigned* hist = (unsigned*)(ws + WS_HIST);
    float*    bsum = (float*)(ws + WS_BSUM);

    vq_init_kernel<<<(KCODES + 255) / 256, 256, 0, stream>>>(hist);
    vq_norm_kernel<<<BN_TOT / 8, 256, 0, stream>>>(z_e, zn, BN_TOT);
    vq_norm_kernel<<<KCODES / 8, 256, 0, stream>>>(cb, en, KCODES);
    vq_argmax_kernel<<<BN_TOT / 128, 128, 0, stream>>>(zn, en, idx, out + OUT_IDX);
    vq_gather_kernel<<<BN_TOT, 256, 0, stream>>>(z_e, cb, idx, out + OUT_ZQ, hist, bsum);
    vq_finalize_kernel<<<1, 256, 0, stream>>>(hist, bsum, out);
}
